// PostProcessCocoGrounding_27994596836147
// MI455X (gfx1250) — matrix operand order repaired and checked by reading, not compiled, not run on hardware
//
#include <hip/hip_runtime.h>

typedef __attribute__((ext_vector_type(16))) _Float16 v16h;
typedef __attribute__((ext_vector_type(8)))  float    v8f;

#define BATCH 64
#define Q     900
#define T     256
#define CDIM  91
#define CP    96          // padded class dim
#define NSEL  300
#define NTILN 6           // 96/16 column tiles per wave
#define NELEM (Q*CP)      // 86400 padded scores per batch
#define CAND  2048        // candidate cap for final sort
#define NBIN  16384       // histogram bins (float bits >> 18)

// Fast sigmoid: v_exp_f32 + v_add + v_rcp_f32. 1-ulp rcp error is far below
// the f16 quantization applied right after (WMMA A operand).
__device__ __forceinline__ _Float16 sigh(float x) {
    return (_Float16)__builtin_amdgcn_rcpf(1.0f + __expf(-x));
}

// ---------------- K0: positive_map (91x256 f32) -> padded (96x256 f16) -------
__global__ __launch_bounds__(256)
void pm_convert_kernel(const float* __restrict__ pm, _Float16* __restrict__ pm16) {
    int i = blockIdx.x * blockDim.x + threadIdx.x;  // over CP*T
    if (i >= CP * T) return;
    int n = i / T, k = i % T;
    float v = (n < CDIM) ? pm[n * T + k] : 0.0f;
    pm16[i] = (_Float16)v;
}

// ---------------- K1: fused sigmoid + WMMA GEMM ------------------------------
// grid = (57 m-tiles, 64 batches), block = 32 (one wave). Each wave computes a
// 16x96 slab of prob[b] with 6 f32 accumulators; 8 k-steps of 16x16x32 WMMA.
__global__ __launch_bounds__(32)
void gemm_wmma_kernel(const float* __restrict__ logits,
                      const _Float16* __restrict__ pm16,
                      float* __restrict__ prob) {
    const int tm   = blockIdx.x;        // 0..56
    const int b    = blockIdx.y;        // 0..63
    const int lane = threadIdx.x;       // 0..31
    const int hid  = lane >> 4;         // half-wave id
    const int l    = lane & 15;

    const int m     = tm * 16 + l;
    const int mrow  = (m < Q) ? m : (Q - 1);      // clamp (padded rows never stored)
    const float* Ab = logits + ((size_t)b * Q + mrow) * T;

    v8f acc[NTILN] = {};

#pragma unroll
    for (int ks = 0; ks < 8; ++ks) {
        // A fragment (contiguous k order, same as B):
        //   lanes 0-15 : elems 0-15 -> K kb..kb+15 (kb = ks*32)
        //   lanes16-31 : same with kb = ks*32 + 16
        const int kb = ks * 32 + hid * 16;
        float4 a0 = *(const float4*)(Ab + kb);
        float4 a1 = *(const float4*)(Ab + kb + 4);
        float4 a2 = *(const float4*)(Ab + kb + 8);
        float4 a3 = *(const float4*)(Ab + kb + 12);
        v16h av;
        av[0]  = sigh(a0.x); av[1]  = sigh(a0.y); av[2]  = sigh(a0.z); av[3]  = sigh(a0.w);
        av[4]  = sigh(a1.x); av[5]  = sigh(a1.y); av[6]  = sigh(a1.z); av[7]  = sigh(a1.w);
        av[8]  = sigh(a2.x); av[9]  = sigh(a2.y); av[10] = sigh(a2.z); av[11] = sigh(a2.w);
        av[12] = sigh(a3.x); av[13] = sigh(a3.y); av[14] = sigh(a3.z); av[15] = sigh(a3.w);

        // B fragment (32x16 16-bit): lane n = l, elems i -> K = khalf + i
        const int khalf = ks * 32 + hid * 16;
#pragma unroll
        for (int t = 0; t < NTILN; ++t) {
            const v16h bv = *(const v16h*)(pm16 + (size_t)(t * 16 + l) * T + khalf);
            acc[t] = __builtin_amdgcn_wmma_f32_16x16x32_f16(
                         false, av, false, bv, (short)0, acc[t], false, false);
        }
    }

    // C/D layout: VGPR r -> M = hid*8 + r, N = l (+16*t)
    const int m0 = tm * 16 + hid * 8;
#pragma unroll
    for (int t = 0; t < NTILN; ++t) {
        const int n = t * 16 + l;
#pragma unroll
        for (int r = 0; r < 8; ++r) {
            const int mm = m0 + r;
            if (mm < Q) {
                float v = (n < CDIM) ? acc[t][r] : -1.0f;   // pad cols can't win top-k
                prob[((size_t)b * Q + mm) * CP + n] = v;
            }
        }
    }
}

// ---------------- K2: per-batch top-300 + box decode -------------------------
__global__ __launch_bounds__(1024)
void topk_kernel(const float* __restrict__ prob,
                 const float* __restrict__ pred_boxes,
                 const float* __restrict__ target_sizes,
                 float* __restrict__ out) {
    __shared__ __align__(16) unsigned int sh_hist[NBIN];   // 64KB, reused for sort keys
    __shared__ unsigned int sh_csum[1024];
    __shared__ unsigned int sh_counter;
    __shared__ unsigned int sh_thresh;

    const int b   = blockIdx.x;
    const int tid = threadIdx.x;
    const float* pb = prob + (size_t)b * NELEM;

    for (int i = tid; i < NBIN; i += 1024) sh_hist[i] = 0u;
    if (tid == 0) sh_counter = 0u;
    __syncthreads();

    // Pass 1: histogram of order-preserving float bits (top 14 bits)
    for (int i = tid; i < NELEM; i += 1024) {
        unsigned int u = __float_as_uint(pb[i]);
        u = (u & 0x80000000u) ? ~u : (u | 0x80000000u);
        atomicAdd(&sh_hist[u >> 18], 1u);
    }
    __syncthreads();

    // Reverse scan for the bin containing rank NSEL
    unsigned int cs = 0;
    for (int j = 0; j < 16; ++j) cs += sh_hist[tid * 16 + j];
    sh_csum[tid] = cs;
    __syncthreads();

    if (tid == 0) {
        unsigned int cum = 0; int chunk = 0;
        for (int t = 1023; t >= 0; --t) {
            if (cum + sh_csum[t] >= NSEL || t == 0) { chunk = t; break; }
            cum += sh_csum[t];
        }
        unsigned int thresh = (unsigned int)chunk * 16u;
        for (int j = 15; j >= 0; --j) {
            unsigned int h = sh_hist[chunk * 16 + j];
            if (cum + h >= NSEL || j == 0) { thresh = (unsigned int)(chunk * 16 + j); break; }
            cum += h;
        }
        sh_thresh = thresh;
    }
    __syncthreads();
    const unsigned int thresh = sh_thresh;

    // Pass 2: compact candidates as 64-bit keys (valbits desc, index asc via ~idx)
    unsigned long long* keys = reinterpret_cast<unsigned long long*>(sh_hist);
    for (int i = tid; i < NELEM; i += 1024) {
        unsigned int u = __float_as_uint(pb[i]);
        u = (u & 0x80000000u) ? ~u : (u | 0x80000000u);
        if ((u >> 18) >= thresh) {
            unsigned int pos = atomicAdd(&sh_counter, 1u);
            if (pos < CAND)
                keys[pos] = ((unsigned long long)u << 32) | (unsigned int)(~(unsigned int)i);
        }
    }
    __syncthreads();
    unsigned int cnt = sh_counter; if (cnt > CAND) cnt = CAND;
    for (int i = tid; i < CAND; i += 1024) if (i >= (int)cnt) keys[i] = 0ull;
    __syncthreads();

    // Bitonic sort, descending, 2048 keys / 1024 threads
    for (int k = 2; k <= CAND; k <<= 1) {
        for (int j = k >> 1; j > 0; j >>= 1) {
#pragma unroll
            for (int base = 0; base < CAND; base += 1024) {
                int i = base + tid;
                int ixj = i ^ j;
                if (ixj > i) {
                    bool desc = ((i & k) == 0);
                    unsigned long long a = keys[i], c2 = keys[ixj];
                    bool sw = desc ? (a < c2) : (a > c2);
                    if (sw) { keys[i] = c2; keys[ixj] = a; }
                }
            }
            __syncthreads();
        }
    }

    // Emit: scores, labels, gathered + scaled boxes
    if (tid < NSEL) {
        unsigned long long key = keys[tid];
        unsigned int u    = (unsigned int)(key >> 32);
        unsigned int idxp = ~((unsigned int)key);
        float val = __uint_as_float(u & 0x7FFFFFFFu);
        int q = (int)(idxp / CP);
        int c = (int)(idxp % CP);
        if (idxp >= NELEM) { q = 0; c = 0; val = 0.0f; }   // defensive

        const int o = b * NSEL + tid;
        out[o] = val;                                 // scores
        out[BATCH * NSEL + o] = (float)c;             // labels (as f32)

        const float* bx = pred_boxes + ((size_t)b * Q + q) * 4;
        float cx = bx[0], cy = bx[1], w = bx[2], h = bx[3];
        float ih = target_sizes[b * 2 + 0];
        float iw = target_sizes[b * 2 + 1];
        float* ob = out + 2 * BATCH * NSEL + (size_t)o * 4;
        ob[0] = (cx - 0.5f * w) * iw;
        ob[1] = (cy - 0.5f * h) * ih;
        ob[2] = (cx + 0.5f * w) * iw;
        ob[3] = (cy + 0.5f * h) * ih;
    }
}

extern "C" void kernel_launch(void* const* d_in, const int* in_sizes, int n_in,
                              void* d_out, int out_size, void* d_ws, size_t ws_size,
                              hipStream_t stream) {
    (void)in_sizes; (void)n_in; (void)out_size; (void)ws_size;
    const float* pred_logits  = (const float*)d_in[0];  // (64,900,256)
    const float* pred_boxes   = (const float*)d_in[1];  // (64,900,4)
    const float* positive_map = (const float*)d_in[2];  // (91,256)
    const float* target_sizes = (const float*)d_in[3];  // (64,2)
    float* out = (float*)d_out;

    _Float16* pm16 = (_Float16*)d_ws;                               // 96*256*2   = 48KB
    float*    prob = (float*)((char*)d_ws + (size_t)CP * T * 2);    // 64*900*96*4 ≈ 22.1MB

    pm_convert_kernel<<<(CP * T + 255) / 256, 256, 0, stream>>>(positive_map, pm16);

    dim3 g1((Q + 15) / 16, BATCH);   // (57, 64)
    gemm_wmma_kernel<<<g1, 32, 0, stream>>>(pred_logits, pm16, prob);

    topk_kernel<<<BATCH, 1024, 0, stream>>>(prob, pred_boxes, target_sizes, out);
}
